// GIN_4836133175915
// MI455X (gfx1250) — matrix operand order repaired and checked by reading, not compiled, not run on hardware
//
#include <hip/hip_runtime.h>
#include <hip/hip_bf16.h>
#include <math.h>

typedef __bf16 bf16_t;
typedef __attribute__((ext_vector_type(8)))  bf16_t v8bf;
typedef __attribute__((ext_vector_type(16))) bf16_t v16bf;
typedef __attribute__((ext_vector_type(8)))  float  v8f;

static __device__ __forceinline__ bf16_t f2bf(float f) {
  union { float f; unsigned u; } v; v.f = f;
  unsigned r = v.u + 0x7FFFu + ((v.u >> 16) & 1u);   // round-to-nearest-even
  union { unsigned short s; bf16_t b; } o; o.s = (unsigned short)(r >> 16);
  return o.b;
}

// ---------------------------------------------------------------------------
// agg = x (rows < N), 0 for padded rows
// ---------------------------------------------------------------------------
__global__ void gin_init_agg(float* __restrict__ agg, const float* __restrict__ x,
                             long long total, long long nvalid) {
  long long i = (long long)blockIdx.x * blockDim.x + threadIdx.x;
  long long stride = (long long)gridDim.x * blockDim.x;
  for (; i < total; i += stride)
    agg[i] = (i < nvalid) ? x[i] : 0.0f;
}

// ---------------------------------------------------------------------------
// agg[dst[e], :] += x[src[e], :]   (one thread per edge x 4-feature chunk)
// ---------------------------------------------------------------------------
__global__ void gin_scatter(float* __restrict__ agg, const float* __restrict__ x,
                            const long long* __restrict__ src,
                            const long long* __restrict__ dst,
                            int F, int lgC, long long E) {
  long long gid = (long long)blockIdx.x * blockDim.x + threadIdx.x;
  long long e = gid >> lgC;
  if (e >= E) return;
  int c = ((int)gid & ((1 << lgC) - 1)) << 2;
  long long s = src[e], d = dst[e];
  const float4 v = *(const float4*)(x + s * (long long)F + c);
  float* o = agg + d * (long long)F + c;
  atomicAdd(o + 0, v.x);
  atomicAdd(o + 1, v.y);
  atomicAdd(o + 2, v.z);
  atomicAdd(o + 3, v.w);
}

// ---------------------------------------------------------------------------
// elementwise f32 -> bf16
// ---------------------------------------------------------------------------
__global__ void gin_f32_to_bf16(bf16_t* __restrict__ o, const float* __restrict__ in,
                                long long n) {
  long long i = (long long)blockIdx.x * blockDim.x + threadIdx.x;
  long long stride = (long long)gridDim.x * blockDim.x;
  for (; i < n; i += stride) o[i] = f2bf(in[i]);
}

// ---------------------------------------------------------------------------
// weight transpose-convert: Wt[n*K + k] = bf16(W[k*N + n])
// ---------------------------------------------------------------------------
__global__ void gin_wt_bf16(bf16_t* __restrict__ o, const float* __restrict__ w,
                            int K, int N) {
  int i = blockIdx.x * blockDim.x + threadIdx.x;
  if (i >= K * N) return;
  int k = i % K, n = i / K;
  o[(size_t)n * K + k] = f2bf(w[(size_t)k * N + n]);
}

// ---------------------------------------------------------------------------
// C = act(A[M x K] @ Bt^T + bias), bf16 inputs, f32 accumulate via WMMA.
// Block: 256 threads = 8 waves; each wave -> 16 rows x 64 cols (4 WMMA tiles).
// A fragment (16x32 bf16): lane<16 -> row M=lane, K {0..15};
//                          lane>=16 -> row M=lane-16, K {16..31}.
// B fragment (32x16 bf16): lane -> col N=(lane&15), K offset (lane>=16)*16.
// C/D tile: VGPR v -> row M = v + 8*(lane>=16), col N = lane&15.
// act: 0=none 1=relu 2=elu 3=sigmoid
// ---------------------------------------------------------------------------
__global__ __launch_bounds__(256) void gin_gemm_wmma(
    const bf16_t* __restrict__ A, const bf16_t* __restrict__ Bt,
    const float* __restrict__ bias, bf16_t* __restrict__ outB,
    float* __restrict__ outF, int K, int Nout, int act, int rowlim) {
  const int lane = threadIdx.x & 31;
  const int wave = threadIdx.x >> 5;
  const int half = lane >> 4;
  const int l16  = lane & 15;
  const int m0 = blockIdx.x * 128 + wave * 16;
  const int n0 = blockIdx.y * 64;
  const int arow = m0 + l16;

  v8f acc[4];
#pragma unroll
  for (int t = 0; t < 4; ++t) acc[t] = (v8f){};

  const bf16_t* aBase = A + (size_t)arow * K + half * 16;
  const bf16_t* bBase = Bt + (size_t)(n0 + l16) * K + half * 16;

  for (int k0 = 0; k0 < K; k0 += 32) {
    v16bf a;
    ((v8bf*)&a)[0] = *(const v8bf*)(aBase + k0);
    ((v8bf*)&a)[1] = *(const v8bf*)(aBase + k0 + 8);
    __builtin_prefetch(aBase + k0 + 32, 0, 3);
#pragma unroll
    for (int t = 0; t < 4; ++t) {
      const v16bf b = *(const v16bf*)(bBase + (size_t)t * 16 * K + k0);
      acc[t] = __builtin_amdgcn_wmma_f32_16x16x32_bf16(
          false, a, false, b, (short)0, acc[t], false, false);
    }
  }

#pragma unroll
  for (int t = 0; t < 4; ++t) {
    const int n = n0 + t * 16 + l16;
    const float bs = bias[n];
#pragma unroll
    for (int v = 0; v < 8; ++v) {
      const int r = m0 + v + half * 8;
      float val = acc[t][v] + bs;
      if (act == 1)      val = fmaxf(val, 0.0f);
      else if (act == 2) val = (val > 0.0f) ? val : (__expf(val) - 1.0f);
      else if (act == 3) val = 1.0f / (1.0f + __expf(-val));
      if (r < rowlim) {
        if (outF) outF[(size_t)r * Nout + n] = val;
        if (outB) outB[(size_t)r * Nout + n] = f2bf(val);
      }
    }
  }
}

// ---------------------------------------------------------------------------
extern "C" void kernel_launch(void* const* d_in, const int* in_sizes, int n_in,
                              void* d_out, int out_size, void* d_ws, size_t ws_size,
                              hipStream_t stream) {
  (void)n_in; (void)out_size; (void)ws_size;
  const int F_IN = 128, HID = 256, EMB = 128, LBL = 64;
  const long long N  = in_sizes[0] / F_IN;
  const long long E  = in_sizes[1] / 2;
  const long long Mp = ((N + 127) / 128) * 128;   // padded rows for 128-row tiles

  const float* x_in = (const float*)d_in[0];
  const long long* ei = (const long long*)d_in[1];
  const long long* src = ei;
  const long long* dst = ei + E;
  const float* w1[3] = {(const float*)d_in[2], (const float*)d_in[6], (const float*)d_in[10]};
  const float* b1[3] = {(const float*)d_in[3], (const float*)d_in[7], (const float*)d_in[11]};
  const float* w2[3] = {(const float*)d_in[4], (const float*)d_in[8], (const float*)d_in[12]};
  const float* b2[3] = {(const float*)d_in[5], (const float*)d_in[9], (const float*)d_in[13]};
  const float* wf = (const float*)d_in[14];
  const float* bfb = (const float*)d_in[15];

  char* ws = (char*)d_ws;
  size_t off = 0;
  auto alloc = [&](size_t bytes) -> void* {
    void* p = ws + off;
    off += (bytes + 255) & ~(size_t)255;
    return p;
  };
  float*  AGG  = (float*)alloc((size_t)Mp * HID * 4);
  float*  X    = (float*)alloc((size_t)Mp * HID * 4);
  bf16_t* AGGb = (bf16_t*)alloc((size_t)Mp * HID * 2);
  bf16_t* H1b  = (bf16_t*)alloc((size_t)Mp * HID * 2);
  bf16_t* Wt1[3]; bf16_t* Wt2[3]; bf16_t* Wtf;
  Wt1[0] = (bf16_t*)alloc((size_t)F_IN * HID * 2);
  Wt2[0] = (bf16_t*)alloc((size_t)HID * HID * 2);
  Wt1[1] = (bf16_t*)alloc((size_t)HID * HID * 2);
  Wt2[1] = (bf16_t*)alloc((size_t)HID * HID * 2);
  Wt1[2] = (bf16_t*)alloc((size_t)HID * HID * 2);
  Wt2[2] = (bf16_t*)alloc((size_t)HID * EMB * 2);
  Wtf    = (bf16_t*)alloc((size_t)EMB * LBL * 2);
  bf16_t* EMBb = AGGb;  // alias: AGG bf16 buffer is free by the time EMB is produced

  // --- one-time weight transpose+convert to bf16 -------------------------
  auto cvtW = [&](bf16_t* o, const float* w, int K, int Nn) {
    int total = K * Nn;
    gin_wt_bf16<<<(total + 255) / 256, 256, 0, stream>>>(o, w, K, Nn);
  };
  cvtW(Wt1[0], w1[0], F_IN, HID);
  cvtW(Wt2[0], w2[0], HID, HID);
  cvtW(Wt1[1], w1[1], HID, HID);
  cvtW(Wt2[1], w2[1], HID, HID);
  cvtW(Wt1[2], w1[2], HID, HID);
  cvtW(Wt2[2], w2[2], HID, EMB);
  cvtW(Wtf,    wf,    EMB, LBL);

  auto gemm = [&](const bf16_t* A, const bf16_t* Bt, const float* bias,
                  bf16_t* oB, float* oF, int K, int Nout, int act, long long rowlim) {
    dim3 grid((unsigned)(Mp / 128), (unsigned)(Nout / 64));
    gin_gemm_wmma<<<grid, 256, 0, stream>>>(A, Bt, bias, oB, oF, K, Nout, act, (int)rowlim);
  };

  const float* xcur = x_in;
  int F = F_IN;
  for (int l = 0; l < 3; ++l) {
    const long long totalF = Mp * (long long)F;
    const long long validF = N * (long long)F;
    gin_init_agg<<<4096, 256, 0, stream>>>(AGG, xcur, totalF, validF);
    const int lgC = (F == 128) ? 5 : 6;             // float4 chunks per edge
    const long long tthreads = E << lgC;
    gin_scatter<<<(unsigned)((tthreads + 255) / 256), 256, 0, stream>>>(
        AGG, xcur, src, dst, F, lgC, E);
    gin_f32_to_bf16<<<4096, 256, 0, stream>>>(AGGb, AGG, totalF);

    gemm(AGGb, Wt1[l], b1[l], H1b, nullptr, F, HID, /*relu*/1, Mp);
    if (l < 2) {
      gemm(H1b, Wt2[l], b2[l], nullptr, X, HID, HID, /*elu*/2, Mp);
      xcur = X; F = HID;
    } else {
      gemm(H1b, Wt2[l], b2[l], EMBb, nullptr, HID, EMB, /*elu*/2, Mp);
    }
  }

  // classifier: sigmoid(EMB @ wf + bf) -> d_out (exact N rows)
  gemm(EMBb, Wtf, bfb, nullptr, (float*)d_out, EMB, LBL, /*sigmoid*/3, N);
}
